// MambaLanguageModel_180388627144
// MI455X (gfx1250) — matrix operand order repaired and checked by reading, not compiled, not run on hardware
//
#include <hip/hip_runtime.h>

// ---------------- model constants ----------------
#define B_SZ   2
#define L_SZ   1024
#define BL     2048          // B*L
#define DMODEL 768
#define DINNER 1536
#define DSTATE 16
#define DTRANK 48
#define DCONV  4
#define VOCAB  32000
#define DBLW   (DTRANK + 2 * DSTATE)   // 80

// ---------------- WMMA types (CDNA5 wave32) ----------------
typedef __attribute__((ext_vector_type(16))) __bf16 v16bf;
typedef __attribute__((ext_vector_type(8)))  float  v8f;

// pack two f32 -> two bf16 in one dword: +0x8000 round-to-nearest, then a
// single v_perm_b32 grabs both high halves (low ushort = a, high ushort = b).
__device__ __forceinline__ unsigned pack2(float a, float b) {
  union { float f; unsigned u; } ua, ub;
  ua.f = a; ub.f = b;
  return __builtin_amdgcn_perm(ub.u + 0x8000u, ua.u + 0x8000u, 0x07060302u);
}

// ---------------- generic bf16-WMMA GEMM ----------------
// C[M,N] (f32, ldc) = A[M,K] (f32, lda) * B[K,N] (f32, ldb) (+ bias[N])
// Block tile 128x128x(K=64), 8 waves (4x2), wave tile 32x64 (2 A-frags x
// 4 B-frags = 8 accumulators -> each B fragment read feeds 2 WMMAs).
// A tile: row-major bf16 in LDS (padded rows). B tile: stored PRE-SWIZZLED in
// WMMA fragment order so each lane's fragment is two contiguous ds_load_b128.
// Requires: K%4==0 (K even for row-pair staging), N%4==0, M%128==0.
#define BLK_M 128
#define BLK_N 128
#define BLK_K 64
#define APAD  8     // As row padding (ushorts) -> 144B row stride, bank-friendly
#define BSTR  12    // Bsw per-lane dword stride (8 data + 4 pad), 48B, 16B-aligned

__global__ __launch_bounds__(256) void gemm_bf16wmma(
    const float* __restrict__ A, int lda,
    const float* __restrict__ Bm, int ldb,
    float* __restrict__ C, int ldc,
    const float* __restrict__ bias,
    int M, int N, int K)
{
  __shared__ unsigned short As[BLK_M][BLK_K + APAD];   // 18 KB
  __shared__ unsigned Bsw[2][8][32][BSTR];             // 24 KB, fragment layout

  const int tid  = threadIdx.x;
  const int lane = tid & 31;
  const int wave = tid >> 5;
  const int wrow = wave >> 1;     // 0..3 : 32-row band
  const int wcol = wave & 1;      // 0..1 : 64-col band
  const int m0 = blockIdx.y * BLK_M;
  const int n0 = blockIdx.x * BLK_N;

  const v8f vzero = {0.f,0.f,0.f,0.f,0.f,0.f,0.f,0.f};
  v8f acc[2][4];
#pragma unroll
  for (int r = 0; r < 2; ++r)
#pragma unroll
    for (int c = 0; c < 4; ++c) acc[r][c] = vzero;

  // A staging: thread -> row ar, 32 cols at acol (8x float4)
  const int ar   = tid >> 1;           // 0..127
  const int acol = (tid & 1) * 32;     // 0/32
  // B staging: thread -> K-row pair (2*brp, 2*brp+1), 16 cols at bct
  const int brp   = tid >> 3;          // 0..31
  const int bct   = (tid & 7) * 16;    // 0..112 (= subtile*16)
  const int bs    = tid & 7;           // 16-col subtile index
  const int bksub = brp >> 4;          // 0/1 : which 32-K half
  const int bkk   = (2 * brp) & 31;    // local k within half-slab (even)
  const int bhl   = (bkk & 16) ? 16 : 0;      // lane-half for khalf
  const int bidx  = (bkk & 15) >> 1;          // dword index (k-pair)

  for (int k0 = 0; k0 < K; k0 += BLK_K) {
    // ---- stage A tile (f32 global -> bf16 LDS, row-major) ----
    {
      const float* pA = A + (size_t)(m0 + ar) * lda + (size_t)(k0 + acol);
      unsigned* dst = (unsigned*)&As[ar][acol];
#pragma unroll
      for (int j = 0; j < 8; ++j) {
        float4 f = make_float4(0.f,0.f,0.f,0.f);
        if (k0 + acol + 4*j < K) f = *(const float4*)(pA + 4*j);
        dst[2*j]   = pack2(f.x, f.y);
        dst[2*j+1] = pack2(f.z, f.w);
      }
    }
    // ---- stage B tile (f32 global -> bf16 LDS, pre-swizzled fragments) ----
    {
      union F16 { float4 q[4]; float f[16]; } r0, r1;
#pragma unroll
      for (int j = 0; j < 4; ++j) {
        r0.q[j] = make_float4(0.f,0.f,0.f,0.f);
        r1.q[j] = make_float4(0.f,0.f,0.f,0.f);
      }
      const int gk = k0 + 2 * brp;
      if (gk < K) {                           // K even -> gk+1 < K too
        const float* p0 = Bm + (size_t)gk * ldb + (size_t)(n0 + bct);
        const float* p1 = p0 + ldb;
#pragma unroll
        for (int j = 0; j < 4; ++j)
          if (n0 + bct + 4*j < N) {           // N % 4 == 0 -> whole chunk in
            r0.q[j] = *(const float4*)(p0 + 4*j);
            r1.q[j] = *(const float4*)(p1 + 4*j);
          }
      }
      // scatter k-pairs into fragment layout: lane = (col&15) + khalf
#pragma unroll
      for (int c = 0; c < 16; ++c)
        Bsw[bksub][bs][c + bhl][bidx] = pack2(r0.f[c], r1.f[c]);
    }
    // L2 prefetch of next K-slab (global_prefetch_b8), overlapped with math
    if (k0 + BLK_K < K) {
      __builtin_prefetch(A + (size_t)(m0 + ar) * lda + (size_t)(k0 + BLK_K + acol), 0, 1);
      if (k0 + BLK_K + 2*brp < K)
        __builtin_prefetch(Bm + (size_t)(k0 + BLK_K + 2*brp) * ldb + (size_t)(n0 + bct), 0, 1);
    }
    __syncthreads();

#pragma unroll
    for (int ks = 0; ks < 2; ++ks) {
      // A fragments (16x32 bf16): lanes0-15 {K0..15}, lanes16-31 {K16..31}
      union { uint4 q[2]; v16bf v; } afrag[2];
#pragma unroll
      for (int r = 0; r < 2; ++r) {
        const uint4* pa =
            (const uint4*)&As[wrow * 32 + r * 16 + (lane & 15)][ks * 32];
        if (lane < 16) { afrag[r].q[0] = pa[0]; afrag[r].q[1] = pa[1]; }
        else           { afrag[r].q[0] = pa[2]; afrag[r].q[1] = pa[3]; }
      }
#pragma unroll
      for (int c = 0; c < 4; ++c) {
        // B fragment: contiguous 32B per lane thanks to staging swizzle
        union { uint4 q[2]; v16bf v; } bfrag;
        const uint4* pb = (const uint4*)&Bsw[ks][wcol * 4 + c][lane][0];
        bfrag.q[0] = pb[0];
        bfrag.q[1] = pb[1];
        acc[0][c] = __builtin_amdgcn_wmma_f32_16x16x32_bf16(
            false, afrag[0].v, false, bfrag.v, (short)0, acc[0][c], false, false);
        acc[1][c] = __builtin_amdgcn_wmma_f32_16x16x32_bf16(
            false, afrag[1].v, false, bfrag.v, (short)0, acc[1][c], false, false);
      }
    }
    __syncthreads();
  }

  // ---- epilogue: 16x16 f32 C layout (VGPR j: lanes0-15 M=j, lanes16-31 M=8+j)
#pragma unroll
  for (int r = 0; r < 2; ++r) {
    const int mbase = m0 + wrow * 32 + r * 16 + ((lane < 16) ? 0 : 8);
#pragma unroll
    for (int c = 0; c < 4; ++c) {
      const int col = n0 + wcol * 64 + c * 16 + (lane & 15);
      if (col < N) {
        const float bv = bias ? bias[col] : 0.f;
#pragma unroll
        for (int j = 0; j < 8; ++j)
          C[(size_t)(mbase + j) * ldc + col] = acc[r][c][j] + bv;
      }
    }
  }
}

// ---------------- embedding gather ----------------
__global__ void embed_kernel(const int* __restrict__ ids,
                             const float* __restrict__ emb,
                             float* __restrict__ x, int total) {
  int i = blockIdx.x * blockDim.x + threadIdx.x;
  if (i >= total) return;
  int t = i / DMODEL, d = i - t * DMODEL;
  x[i] = emb[(size_t)ids[t] * DMODEL + d];
}

// ---------------- causal depthwise conv (K=4) + SiLU ----------------
__global__ void conv_silu_kernel(const float* __restrict__ xz,
                                 const float* __restrict__ cw,
                                 const float* __restrict__ cb,
                                 float* __restrict__ xa, int total) {
  int i = blockIdx.x * blockDim.x + threadIdx.x;
  if (i >= total) return;
  int d = i % DINNER;
  int l = (i / DINNER) % L_SZ;
  int b = i / (DINNER * L_SZ);
  const float* xi = xz + (size_t)b * L_SZ * (2 * DINNER) + d;  // xi half of xz rows
  float s = cb[d];
#pragma unroll
  for (int j = 0; j < DCONV; ++j) {
    int ls = l - (DCONV - 1) + j;
    if (ls >= 0) s += cw[d * DCONV + j] * xi[(size_t)ls * (2 * DINNER)];
  }
  xa[i] = s / (1.f + __expf(-s));     // SiLU
}

// ---------------- dt bias + softplus (in place) ----------------
__global__ void dt_softplus_kernel(float* __restrict__ dt,
                                   const float* __restrict__ bias, int total) {
  int i = blockIdx.x * blockDim.x + threadIdx.x;
  if (i >= total) return;
  float x = dt[i] + bias[i % DINNER];
  dt[i] = (x > 20.f) ? x : __logf(1.f + __expf(x));
}

// ---------------- selective scan (+D skip, +SiLU(z) gate) ----------------
// One lane owns one (b,d) channel; h[16] lives in VGPRs; per-step B/C (32
// floats) broadcast through double-buffered LDS (one barrier per step).
__global__ __launch_bounds__(256) void scan_kernel(
    const float* __restrict__ dbl,   // BL x 80   (dt|B|C)
    const float* __restrict__ dt,    // BL x 1536 (post-softplus)
    const float* __restrict__ xa,    // BL x 1536
    const float* __restrict__ xz,    // BL x 3072 (z at col 1536+)
    const float* __restrict__ A_log, // 1536 x 16
    const float* __restrict__ Dp,    // 1536
    float* __restrict__ y)           // BL x 1536
{
  __shared__ float sBC[2][2 * DSTATE];
  const int bpb = DINNER / 256;                    // blocks per batch = 6
  const int b = blockIdx.x / bpb;
  const int d = (blockIdx.x % bpb) * 256 + threadIdx.x;
  float Arow[DSTATE], h[DSTATE];
#pragma unroll
  for (int n = 0; n < DSTATE; ++n) {
    Arow[n] = -__expf(A_log[d * DSTATE + n]);
    h[n] = 0.f;
  }
  const float Dd = Dp[d];
  for (int l = 0; l < L_SZ; ++l) {
    const int t = b * L_SZ + l;
    const int buf = l & 1;
    if (threadIdx.x < 2 * DSTATE)
      sBC[buf][threadIdx.x] = dbl[(size_t)t * DBLW + DTRANK + threadIdx.x];
    __syncthreads();
    const float dtv = dt[(size_t)t * DINNER + d];
    const float xv  = xa[(size_t)t * DINNER + d];
    float acc = 0.f;
#pragma unroll
    for (int n = 0; n < DSTATE; ++n) {
      h[n] = __expf(dtv * Arow[n]) * h[n] + dtv * sBC[buf][n] * xv;
      acc += h[n] * sBC[buf][DSTATE + n];
    }
    const float zv = xz[(size_t)t * (2 * DINNER) + DINNER + d];
    y[(size_t)t * DINNER + d] = (acc + xv * Dd) * (zv / (1.f + __expf(-zv)));
  }
}

// ---------------- host orchestration ----------------
extern "C" void kernel_launch(void* const* d_in, const int* in_sizes, int n_in,
                              void* d_out, int out_size, void* d_ws, size_t ws_size,
                              hipStream_t stream) {
  (void)in_sizes; (void)n_in; (void)out_size; (void)ws_size;
  const int*   ids   = (const int*)  d_in[0];
  const float* emb   = (const float*)d_in[1];
  const float* w_in  = (const float*)d_in[2];   // 768 x 3072
  const float* cw    = (const float*)d_in[3];   // 1536 x 4
  const float* cbias = (const float*)d_in[4];
  const float* w_xp  = (const float*)d_in[5];   // 1536 x 80
  const float* w_dt  = (const float*)d_in[6];   // 48 x 1536
  const float* b_dt  = (const float*)d_in[7];
  const float* A_log = (const float*)d_in[8];   // 1536 x 16
  const float* Dp    = (const float*)d_in[9];
  const float* w_out = (const float*)d_in[10];  // 1536 x 768
  const float* w_lm  = (const float*)d_in[11];  // 768 x 32000
  const float* b_lm  = (const float*)d_in[12];
  float* out = (float*)d_out;

  size_t off = 0;
  auto alloc = [&](size_t elems) -> float* {
    float* p = (float*)((char*)d_ws + off);
    off += ((elems * sizeof(float)) + 255u) & ~(size_t)255u;
    return p;
  };
  float* x   = alloc((size_t)BL * DMODEL);      //  6.3 MB
  float* xz  = alloc((size_t)BL * 2 * DINNER);  // 25.2 MB
  float* xa  = alloc((size_t)BL * DINNER);      // 12.6 MB
  float* dbl = alloc((size_t)BL * DBLW);        //  0.7 MB
  float* dtf = alloc((size_t)BL * DINNER);      // 12.6 MB
  float* y   = alloc((size_t)BL * DINNER);      // 12.6 MB
  float* xo  = alloc((size_t)BL * DMODEL);      //  6.3 MB

  // 1) embedding gather
  embed_kernel<<<(BL * DMODEL + 255) / 256, 256, 0, stream>>>(ids, emb, x, BL * DMODEL);
  // 2) in_proj: (2048x768)@(768x3072) -> xz
  gemm_bf16wmma<<<dim3((2 * DINNER + BLK_N - 1) / BLK_N, BL / BLK_M), 256, 0, stream>>>(
      x, DMODEL, w_in, 2 * DINNER, xz, 2 * DINNER, nullptr, BL, 2 * DINNER, DMODEL);
  // 3) causal depthwise conv + SiLU -> xa
  conv_silu_kernel<<<(BL * DINNER + 255) / 256, 256, 0, stream>>>(xz, cw, cbias, xa, BL * DINNER);
  // 4) x_proj: (2048x1536)@(1536x80) -> dbl
  gemm_bf16wmma<<<dim3(1, BL / BLK_M), 256, 0, stream>>>(
      xa, DINNER, w_xp, DBLW, dbl, DBLW, nullptr, BL, DBLW, DINNER);
  // 5) dt_proj: dt slice of dbl (lda=80, K=48) @ (48x1536) -> dtf
  gemm_bf16wmma<<<dim3((DINNER + BLK_N - 1) / BLK_N, BL / BLK_M), 256, 0, stream>>>(
      dbl, DBLW, w_dt, DINNER, dtf, DINNER, nullptr, BL, DINNER, DTRANK);
  // 6) softplus(dt + bias)
  dt_softplus_kernel<<<(BL * DINNER + 255) / 256, 256, 0, stream>>>(dtf, b_dt, BL * DINNER);
  // 7) selective scan + D skip + SiLU(z) gate -> y
  scan_kernel<<<B_SZ * (DINNER / 256), 256, 0, stream>>>(dbl, dtf, xa, xz, A_log, Dp, y);
  // 8) out_proj: (2048x1536)@(1536x768) -> xo
  gemm_bf16wmma<<<dim3((DMODEL + BLK_N - 1) / BLK_N, BL / BLK_M), 256, 0, stream>>>(
      y, DINNER, w_out, DMODEL, xo, DMODEL, nullptr, BL, DMODEL, DINNER);
  // 9) lm_head: (2048x768)@(768x32000) + bias -> logits (f32 out)
  gemm_bf16wmma<<<dim3((VOCAB + BLK_N - 1) / BLK_N, BL / BLK_M), 256, 0, stream>>>(
      xo, DMODEL, w_lm, VOCAB, out, VOCAB, b_lm, BL, VOCAB, DMODEL);
}
